// RNNNode_56856777064872
// MI455X (gfx1250) — matrix operand order repaired and checked by reading, not compiled, not run on hardware
//
#include <hip/hip_runtime.h>

#define BATCH   65536
#define D_IN    128
#define D_HID   512
#define D_OUT   128

typedef __attribute__((ext_vector_type(16))) _Float16 v16h;
typedef __attribute__((ext_vector_type(8)))  _Float16 v8h;
typedef __attribute__((ext_vector_type(4)))  _Float16 v4h;
typedef __attribute__((ext_vector_type(8)))  float    v8f;
typedef __attribute__((ext_vector_type(4)))  float    v4f;

// ---------------- softmax over a[512] -> decay[512] ----------------
__global__ void k_softmax(const float* __restrict__ a, float* __restrict__ decay) {
    __shared__ float red[512];
    const int t = threadIdx.x;
    float v = a[t];
    red[t] = v;
    __syncthreads();
    for (int s = 256; s > 0; s >>= 1) {
        if (t < s) red[t] = fmaxf(red[t], red[t + s]);
        __syncthreads();
    }
    float m = red[0];
    __syncthreads();
    float e = expf(v - m);
    red[t] = e;
    __syncthreads();
    for (int s = 256; s > 0; s >>= 1) {
        if (t < s) red[t] += red[t + s];
        __syncthreads();
    }
    decay[t] = e / red[0];
}

// ------------- transpose + f32->f16 convert: dst[C][R] = src[R][C] -------------
__global__ void k_transpose_f16(const float* __restrict__ src, _Float16* __restrict__ dst,
                                int R, int C) {
    int i = blockIdx.x * 256 + threadIdx.x;
    if (i >= R * C) return;
    int r = i / C;
    int c = i % C;
    dst[c * R + r] = (_Float16)src[i];
}

// ---------------- fragment loaders ----------------
// B fragment (32x16 f16) from K-major (transposed) matrix, row length ld:
// lane L holds column n0+(L&15), K = kk*32+(L>>4)*16 .. +15 -> contiguous 32B load.
__device__ __forceinline__ v16h load_bfrag(const _Float16* __restrict__ T, int ld,
                                           int n0, int kk, int lane) {
    int n  = n0 + (lane & 15);
    int kb = kk * 32 + ((lane >> 4) << 4);
    return *(const v16h*)(T + n * ld + kb);
}

// A fragment (16x32 f16) from row-major f16 tile S[16][ldk] in LDS:
// lane L: m = L&15, kb = kk*32 + (L>>4)*16; elems 0..7 = K kb..kb+7,
// elems 8..15 = K kb+8..kb+15 -> two ds_load_b128.
__device__ __forceinline__ v16h load_afrag(const _Float16* S, int ldk, int kk, int lane) {
    int m  = lane & 15;
    int kb = kk * 32 + ((lane >> 4) << 4);
    const _Float16* p = S + m * ldk + kb;
    v8h lo = *(const v8h*)(p);
    v8h hi = *(const v8h*)(p + 8);
    return __builtin_shufflevector(lo, hi, 0, 1, 2, 3, 4, 5, 6, 7,
                                           8, 9, 10, 11, 12, 13, 14, 15);
}

// ---------------- fused: x_new = decay*x + u@b ; y = x_new@c ----------------
// 256 threads = 8 waves (wave32). Workgroup owns 64 batch rows (4 row-tiles of 16).
// GEMM1: wave w owns hidden columns [w*64, w*64+64)  (4 n-tiles x 4 k-steps)
// GEMM2: wave w owns output columns [w*16, w*16+16)  (1 n-tile  x 16 k-steps)
__launch_bounds__(256)
__global__ void k_fused(const float* __restrict__ x, const float* __restrict__ u,
                        const float* __restrict__ decay,
                        const _Float16* __restrict__ bT, const _Float16* __restrict__ cT,
                        float* __restrict__ xnew, float* __restrict__ y) {
    __shared__ _Float16 uh[16 * D_IN];   // 4 KB: u row-tile, f16
    __shared__ _Float16 xh[16 * D_HID];  // 16 KB: x_new row-tile, f16

    const int lane = threadIdx.x & 31;
    const int w    = threadIdx.x >> 5;
    const int blockRow0 = blockIdx.x * 64;
    const int mb   = (lane >> 4) << 3;   // C/D layout: this lane's row base in tile
    const int nlo  = lane & 15;          // C/D layout: this lane's column in tile

    // ---- batch-invariant data -> registers ----
    v16h B1[4][4];                       // GEMM1 B fragments (bT, ld = D_IN)
#pragma unroll
    for (int j = 0; j < 4; ++j)
#pragma unroll
        for (int kk = 0; kk < 4; ++kk)
            B1[j][kk] = load_bfrag(bT, D_IN, w * 64 + j * 16, kk, lane);

    v16h B2[16];                         // GEMM2 B fragments (cT, ld = D_HID)
#pragma unroll
    for (int kk = 0; kk < 16; ++kk)
        B2[kk] = load_bfrag(cT, D_HID, w * 16, kk, lane);

    float dj[4];
#pragma unroll
    for (int j = 0; j < 4; ++j)
        dj[j] = decay[w * 64 + j * 16 + nlo];

    // ---- row-tile-invariant pointers (advance via immediate offsets) ----
    const float* xp[4];
    float*       xnp[4];
    _Float16*    xhp[4];
#pragma unroll
    for (int j = 0; j < 4; ++j) {
        int col = w * 64 + j * 16 + nlo;
        xp[j]  = x    + (blockRow0 + mb) * D_HID + col;
        xnp[j] = xnew + (blockRow0 + mb) * D_HID + col;
        xhp[j] = xh + mb * D_HID + col;
    }
    float* yp = y + (blockRow0 + mb) * D_OUT + w * 16 + nlo;

    for (int rt = 0; rt < 4; ++rt) {
        const int row0 = blockRow0 + rt * 16;
        const int ro   = rt * 16 * D_HID;   // f32 element offset into x / xnew
        const int yo   = rt * 16 * D_OUT;

        // stage u row-tile into LDS as f16 (b128 loads, NT, ds_store_b64)
        {
            const v4f* us = (const v4f*)(u + row0 * D_IN);
            for (int i = threadIdx.x; i < (16 * D_IN) / 4; i += 256) {
                v4f v = __builtin_nontemporal_load(us + i);
                v4h h = { (_Float16)v.x, (_Float16)v.y, (_Float16)v.z, (_Float16)v.w };
                *(v4h*)(uh + i * 4) = h;
            }
        }
        __syncthreads();

        // shared A fragments over K = 0..127
        v16h A1[4];
#pragma unroll
        for (int kk = 0; kk < 4; ++kk)
            A1[kk] = load_afrag(uh, D_IN, kk, lane);

        // ---- GEMM1 + elementwise fuse ----
#pragma unroll
        for (int j = 0; j < 4; ++j) {
            // batch all 8 x loads for this tile; they overlap the WMMA chain below
            float xv[8];
#pragma unroll
            for (int r = 0; r < 8; ++r)
                xv[r] = __builtin_nontemporal_load(xp[j] + ro + r * D_HID);

            v8f acc = {};
#pragma unroll
            for (int kk = 0; kk < 4; ++kk)
                acc = __builtin_amdgcn_wmma_f32_16x16x32_f16(
                    false, A1[kk], false, B1[j][kk], (short)0, acc, false, false);

#pragma unroll
            for (int r = 0; r < 8; ++r) {
                float xn = fmaf(dj[j], xv[r], acc[r]);
                __builtin_nontemporal_store(xn, xnp[j] + ro + r * D_HID);
                xhp[j][r * D_HID] = (_Float16)xn;   // f16 copy for GEMM2
            }
        }
        __syncthreads();

        // ---- GEMM2: y tile = x_new @ c ----
        v8f acc2 = {};
#pragma unroll
        for (int kk = 0; kk < 16; ++kk) {
            v16h A2 = load_afrag(xh, D_HID, kk, lane);
            acc2 = __builtin_amdgcn_wmma_f32_16x16x32_f16(
                false, A2, false, B2[kk], (short)0, acc2, false, false);
        }
#pragma unroll
        for (int r = 0; r < 8; ++r)
            __builtin_nontemporal_store(acc2[r], yp + yo + r * D_OUT);

        __syncthreads();   // protect uh/xh before next row-tile
    }
}

extern "C" void kernel_launch(void* const* d_in, const int* in_sizes, int n_in,
                              void* d_out, int out_size, void* d_ws, size_t ws_size,
                              hipStream_t stream) {
    // setup_inputs() order: x, u, a, b, c  (all float32)
    const float* x = (const float*)d_in[0];
    const float* u = (const float*)d_in[1];
    const float* a = (const float*)d_in[2];
    const float* b = (const float*)d_in[3];
    const float* c = (const float*)d_in[4];

    float* out  = (float*)d_out;
    float* xnew = out;                              // [BATCH, D_HID]
    float* y    = out + (size_t)BATCH * D_HID;      // [BATCH, D_OUT]

    // workspace layout: decay f32 [512] | bT f16 [512][128] | cT f16 [128][512]
    float*    decay = (float*)d_ws;
    _Float16* bT    = (_Float16*)((char*)d_ws + 2048);
    _Float16* cT    = (_Float16*)((char*)d_ws + 2048 + (size_t)D_IN * D_HID * 2);

    k_softmax<<<1, 512, 0, stream>>>(a, decay);
    k_transpose_f16<<<(D_IN * D_HID + 255) / 256, 256, 0, stream>>>(b, bT, D_IN, D_HID);
    k_transpose_f16<<<(D_HID * D_OUT + 255) / 256, 256, 0, stream>>>(c, cT, D_HID, D_OUT);
    k_fused<<<BATCH / 64, 256, 0, stream>>>(x, u, decay, bT, cT, xnew, y);
}
